// MMA_NSE_22651657519409
// MI455X (gfx1250) — matrix operand order repaired and checked by reading, not compiled, not run on hardware
//
#include <hip/hip_runtime.h>

// ---------------------------------------------------------------------------
// CDNA5 (gfx1250) implementation of the NTM-style memory RNN.
// R=512, B=16, K=2 memories, N=256 slots, T=64 steps.
// Dense GEMMs run on v_wmma_f32_16x16x32_f16 (f16 in, f32 accumulate).
// Attention / softmax / memory update stay in f32 VALU (L2-resident,
// bandwidth-bound; 192MB L2 holds all weights + Mq).
// ---------------------------------------------------------------------------

typedef __attribute__((ext_vector_type(16))) _Float16 v16h;
typedef __attribute__((ext_vector_type(8)))  float    v8f;

#define RNN    512
#define BATCH  16
#define NSLOT  256
#define KMEM   2
#define TSTEPS 64

// ---------------- WMMA fragment loaders (wave32, 16x16x32 f16) -------------
// A fragment as loaded here (contiguous K, not AMD's table order): lanes 0-15 -> M=0..15 with
// K = 0..15; lanes 16-31 -> same M rows, K = 16..31 (matches B).
// Halves are packed 2-per-VGPR with contiguous K pairs.
__device__ __forceinline__ v16h load_a_frag(const _Float16* aT, int stride,
                                            int kk, int lane) {
  int m  = lane & 15;
  int kb = (lane & 16) ? 16 : 0;
  const _Float16* row = aT + m * stride + kk;
  v16h a;
#pragma unroll
  for (int j = 0; j < 8; ++j) {
    int kp = 2 * j + kb;
    a[2 * j]     = row[kp];
    a[2 * j + 1] = row[kp + 1];
  }
  return a;
}

// B fragment as loaded here (contiguous K, same as A): lanes 0-15 -> N=lane, K=0..15 (2 K-rows per
// VGPR); lanes 16-31 -> N=lane-16, K=16..31.  W is stored row-major
// [out_unit][k], so each lane reads 16 contiguous halves of row (j0+n).
__device__ __forceinline__ v16h load_b_frag(const _Float16* W, int ldw,
                                            int j0, int kk, int lane) {
  int n  = lane & 15;
  int kb = (lane & 16) ? 16 : 0;
  return *(const v16h*)(W + (size_t)(j0 + n) * ldw + kk + kb);
}

// ---------------- generic skinny GEMM: out[16][N] = A[16][K] @ W^T + bias --
// One wave per 16-column tile. A staged through LDS (shows ds path), B read
// directly from global f16 weights. Two K-segments (e.g. Wih | Whh).
__global__ void gemm_kernel(const _Float16* __restrict__ A, int Ktot,
                            const _Float16* __restrict__ W0, int ldw0, int klen0,
                            const _Float16* __restrict__ W1, int ldw1, int klen1,
                            const float* __restrict__ bias,
                            float* __restrict__ out, int N) {
  __shared__ _Float16 aT[16 * 1536];   // up to 48KB
  int lane = threadIdx.x;              // 32 threads = 1 wave
  int j0   = blockIdx.x * 16;

  { // stage A (f16, contiguous) into LDS as uint4 chunks
    const uint4* s = (const uint4*)A;
    uint4*       d = (uint4*)aT;
    int n4 = (16 * Ktot) >> 3;
    for (int i = lane; i < n4; i += 32) d[i] = s[i];
  }
  __syncthreads();

  v8f acc = {0.f, 0.f, 0.f, 0.f, 0.f, 0.f, 0.f, 0.f};
  for (int kk = 0; kk < klen0; kk += 32) {
    v16h a = load_a_frag(aT, Ktot, kk, lane);
    v16h b = load_b_frag(W0, ldw0, j0, kk, lane);
    acc = __builtin_amdgcn_wmma_f32_16x16x32_f16(false, a, false, b,
                                                 (short)0, acc, false, false);
  }
  for (int kk = 0; kk < klen1; kk += 32) {
    v16h a = load_a_frag(aT, Ktot, klen0 + kk, lane);
    v16h b = load_b_frag(W1, ldw1, j0, kk, lane);
    acc = __builtin_amdgcn_wmma_f32_16x16x32_f16(false, a, false, b,
                                                 (short)0, acc, false, false);
  }

  // C/D layout: VGPR r, lanes 0-15 -> M=r, N=lane; lanes 16-31 -> M=r+8.
  int n = lane & 15;
  int boff = (lane & 16) ? 8 : 0;
  int j = j0 + n;
  float bj = bias ? bias[j] : 0.f;
#pragma unroll
  for (int r = 0; r < 8; ++r)
    out[(size_t)(r + boff) * N + j] = acc[r] + bj;
}

// ---------------- build f16 A-matrix from up to 3 f32 segments -------------
__global__ void build3_kernel(_Float16* __restrict__ dst,
                              const float* __restrict__ s0, int l0,
                              const float* __restrict__ s1, int l1,
                              const float* __restrict__ s2, int l2) {
  int L = l0 + l1 + l2;
  int total = 16 * L;
  for (int i = blockIdx.x * blockDim.x + threadIdx.x; i < total;
       i += gridDim.x * blockDim.x) {
    int row = i / L, col = i % L;
    float v;
    if (col < l0)            v = s0[row * l0 + col];
    else if (col < l0 + l1)  v = s1[row * l1 + (col - l0)];
    else                     v = s2[row * l2 + (col - l0 - l1)];
    dst[i] = (_Float16)v;
  }
}

// ---------------- LSTM gate activation (i,f,g,o layout) --------------------
__global__ void lstm_act_kernel(const float* __restrict__ gates,
                                const float* __restrict__ c_in,
                                float* __restrict__ c_out,
                                float* __restrict__ h_out,
                                float* __restrict__ extra) {
  int i = blockIdx.x * blockDim.x + threadIdx.x;   // 16*512
  if (i >= BATCH * RNN) return;
  int b = i >> 9, j = i & (RNN - 1);
  const float* g = gates + (size_t)b * 4 * RNN;
  float iv = 1.f / (1.f + __expf(-g[j]));
  float fv = 1.f / (1.f + __expf(-g[RNN + j]));
  float gv = tanhf(g[2 * RNN + j]);
  float ov = 1.f / (1.f + __expf(-g[3 * RNN + j]));
  float c  = fv * c_in[i] + iv * gv;
  float h  = ov * tanhf(c);
  c_out[i] = c;
  h_out[i] = h;
  if (extra) extra[i] = h;
}

// ---------------- memory read: sim -> softmax(z) -> m ----------------------
// One block per (k,b): 256 threads.
__global__ void mem_read_kernel(const float* __restrict__ Mq,
                                const float* __restrict__ hr,
                                float* __restrict__ z,
                                float* __restrict__ m) {
  int kb = blockIdx.x;                 // k*16 + b
  int b  = kb & 15;
  const float* Mb = Mq + (size_t)kb * NSLOT * RNN;
  __shared__ float hr_sh[RNN];
  __shared__ float z_sh[NSLOT];
  __shared__ float red[32];
  int tid = threadIdx.x;
  hr_sh[tid]        = hr[b * RNN + tid];
  hr_sh[tid + 256]  = hr[b * RNN + tid + 256];
  __syncthreads();

  int lane = tid & 31, rowg = tid >> 5;   // 8 waves, 8 slot-rows at a time
  for (int n0 = 0; n0 < NSLOT; n0 += 8) {
    int n = n0 + rowg;
    const float* Mr = Mb + (size_t)n * RNN;
    float p = 0.f;
    for (int r = lane; r < RNN; r += 32) p += Mr[r] * hr_sh[r];
    for (int off = 16; off > 0; off >>= 1) p += __shfl_xor(p, off, 32);
    if (lane == 0) z_sh[n] = p;
  }
  __syncthreads();

  // softmax over 256 slots
  float v  = z_sh[tid];
  float mx = v;
  for (int off = 16; off > 0; off >>= 1) mx = fmaxf(mx, __shfl_xor(mx, off, 32));
  if (lane == 0) red[rowg] = mx;
  __syncthreads();
  if (tid == 0) { float t = red[0]; for (int i = 1; i < 8; ++i) t = fmaxf(t, red[i]); red[16] = t; }
  __syncthreads();
  mx = red[16];
  float e = __expf(v - mx);
  float s = e;
  for (int off = 16; off > 0; off >>= 1) s += __shfl_xor(s, off, 32);
  if (lane == 0) red[8 + rowg] = s;
  __syncthreads();
  if (tid == 0) { float t = 0.f; for (int i = 0; i < 8; ++i) t += red[8 + i]; red[17] = 1.f / t; }
  __syncthreads();
  float zn = e * red[17];
  z_sh[tid] = zn;
  z[(size_t)kb * NSLOT + tid] = zn;
  __syncthreads();

  // m[r] = sum_n z[n] * M[n][r]   (threads own r = tid, tid+256; coalesced)
  float m0 = 0.f, m1 = 0.f;
  for (int n = 0; n < NSLOT; ++n) {
    const float* Mr = Mb + (size_t)n * RNN;
    __builtin_prefetch(Mr + RNN, 0, 1);
    float zv = z_sh[n];
    m0 += zv * Mr[tid];
    m1 += zv * Mr[tid + 256];
  }
  m[(size_t)kb * RNN + tid]       = m0;
  m[(size_t)kb * RNN + tid + 256] = m1;
}

// ---------------- row softmax for comp (16 rows x 512) ---------------------
__global__ void row_softmax_kernel(const float* __restrict__ logits,
                                   float* __restrict__ out) {
  int tid = threadIdx.x;               // 512 threads = 16 waves, wave = row
  int wave = tid >> 5, lane = tid & 31;
  const float* row = logits + (size_t)wave * RNN;
  float mx = -3.4e38f;
  for (int j = lane; j < RNN; j += 32) mx = fmaxf(mx, row[j]);
  for (int off = 16; off > 0; off >>= 1) mx = fmaxf(mx, __shfl_xor(mx, off, 32));
  float s = 0.f;
  for (int j = lane; j < RNN; j += 32) s += __expf(row[j] - mx);
  for (int off = 16; off > 0; off >>= 1) s += __shfl_xor(s, off, 32);
  float inv = 1.f / s;
  for (int j = lane; j < RNN; j += 32)
    out[(size_t)wave * RNN + j] = __expf(row[j] - mx) * inv;
}

// ---------------- NTM update: Mq = (1 - z) * (Mq + hw) ---------------------
__global__ void mem_update_kernel(float* __restrict__ Mq,
                                  const float* __restrict__ z,
                                  const float* __restrict__ hw) {
  size_t i = (size_t)blockIdx.x * blockDim.x + threadIdx.x;   // 1M float4s
  size_t e = i * 4;
  int r = (int)(e & (RNN - 1));
  size_t nbk = e >> 9;                       // k*B*N + b*N + n
  int b = (int)((nbk >> 8) & 15);
  float er = 1.0f - z[nbk];
  float4 hv = *(const float4*)(hw + (size_t)b * RNN + r);
  float4 mv = *(float4*)(Mq + e);
  mv.x = er * (mv.x + hv.x);
  mv.y = er * (mv.y + hv.y);
  mv.z = er * (mv.z + hv.z);
  mv.w = er * (mv.w + hv.w);
  *(float4*)(Mq + e) = mv;
}

// ---------------- one-time converts --------------------------------------
__global__ void cvt_f16_kernel(const float* __restrict__ s,
                               _Float16* __restrict__ d, int n) {
  int i = blockIdx.x * blockDim.x + threadIdx.x;
  if (i < n) d[i] = (_Float16)s[i];
}
__global__ void add_vec_kernel(const float* __restrict__ a,
                               const float* __restrict__ b,
                               float* __restrict__ o, int n) {
  int i = blockIdx.x * blockDim.x + threadIdx.x;
  if (i < n) o[i] = a[i] + b[i];
}

// ===========================================================================
extern "C" void kernel_launch(void* const* d_in, const int* in_sizes, int n_in,
                              void* d_out, int out_size, void* d_ws, size_t ws_size,
                              hipStream_t stream) {
  (void)in_sizes; (void)n_in; (void)out_size; (void)ws_size;
  const float* emb      = (const float*)d_in[0];   // [64,16,512]
  const float* hr0      = (const float*)d_in[1];
  const float* cr0      = (const float*)d_in[2];
  const float* hw0      = (const float*)d_in[3];
  const float* cw0      = (const float*)d_in[4];
  const float* init_out = (const float*)d_in[5];   // [16,512]
  const float* Min      = (const float*)d_in[6];   // [2,16,256,512]
  const float* Wih_r    = (const float*)d_in[7];   // [2048,1024]
  const float* Whh_r    = (const float*)d_in[8];   // [2048,512]
  const float* bih_r    = (const float*)d_in[9];
  const float* bhh_r    = (const float*)d_in[10];
  const float* Wc       = (const float*)d_in[11];  // [512,1536]
  const float* bc       = (const float*)d_in[12];
  const float* Wih_w    = (const float*)d_in[13];  // [2048,512]
  const float* Whh_w    = (const float*)d_in[14];
  const float* bih_w    = (const float*)d_in[15];
  const float* bhh_w    = (const float*)d_in[16];
  float* out = (float*)d_out;

  // ---- workspace carve (aligned 256B) ----
  char* w = (char*)d_ws;
  auto carve = [&](size_t bytes) -> char* {
    char* p = w; w += (bytes + 255) & ~(size_t)255; return p;
  };
  const size_t MQ_ELEMS = (size_t)KMEM * BATCH * NSLOT * RNN;       // 4194304
  float*    Mq       = (float*)carve(MQ_ELEMS * 4);
  float*    hrA      = (float*)carve(BATCH * RNN * 4);
  float*    hrB      = (float*)carve(BATCH * RNN * 4);
  float*    hwA      = (float*)carve(BATCH * RNN * 4);
  float*    hwB      = (float*)carve(BATCH * RNN * 4);
  float*    cr       = (float*)carve(BATCH * RNN * 4);
  float*    cw       = (float*)carve(BATCH * RNN * 4);
  float*    gates_ws = (float*)carve(BATCH * 4 * RNN * 4);
  float*    logit_ws = (float*)carve(BATCH * RNN * 4);
  float*    comp_ws  = (float*)carve(BATCH * RNN * 4);
  float*    m_ws     = (float*)carve((size_t)KMEM * BATCH * RNN * 4);
  float*    z_ws     = (float*)carve((size_t)KMEM * BATCH * NSLOT * 4);
  float*    br       = (float*)carve(4 * RNN * 4);
  float*    bw       = (float*)carve(4 * RNN * 4);
  _Float16* xr_h     = (_Float16*)carve((size_t)BATCH * 1536 * 2);
  _Float16* cat_h    = (_Float16*)carve((size_t)BATCH * 1536 * 2);
  _Float16* xw_h     = (_Float16*)carve((size_t)BATCH * 1024 * 2);
  _Float16* Wih_r_h  = (_Float16*)carve((size_t)2048 * 1024 * 2);
  _Float16* Whh_r_h  = (_Float16*)carve((size_t)2048 * 512 * 2);
  _Float16* Wc_h     = (_Float16*)carve((size_t)512 * 1536 * 2);
  _Float16* Wih_w_h  = (_Float16*)carve((size_t)2048 * 512 * 2);
  _Float16* Whh_w_h  = (_Float16*)carve((size_t)2048 * 512 * 2);

  // ---- init state + memory ----
  hipMemcpyAsync(Mq,  Min, MQ_ELEMS * 4,        hipMemcpyDeviceToDevice, stream);
  hipMemcpyAsync(hrA, hr0, BATCH * RNN * 4,     hipMemcpyDeviceToDevice, stream);
  hipMemcpyAsync(cr,  cr0, BATCH * RNN * 4,     hipMemcpyDeviceToDevice, stream);
  hipMemcpyAsync(hwA, hw0, BATCH * RNN * 4,     hipMemcpyDeviceToDevice, stream);
  hipMemcpyAsync(cw,  cw0, BATCH * RNN * 4,     hipMemcpyDeviceToDevice, stream);

  // ---- one-time weight converts (f32 -> f16) + bias sums ----
  cvt_f16_kernel<<<8192, 256, 0, stream>>>(Wih_r, Wih_r_h, 2048 * 1024);
  cvt_f16_kernel<<<4096, 256, 0, stream>>>(Whh_r, Whh_r_h, 2048 * 512);
  cvt_f16_kernel<<<3072, 256, 0, stream>>>(Wc,    Wc_h,    512 * 1536);
  cvt_f16_kernel<<<4096, 256, 0, stream>>>(Wih_w, Wih_w_h, 2048 * 512);
  cvt_f16_kernel<<<4096, 256, 0, stream>>>(Whh_w, Whh_w_h, 2048 * 512);
  add_vec_kernel<<<8, 256, 0, stream>>>(bih_r, bhh_r, br, 4 * RNN);
  add_vec_kernel<<<8, 256, 0, stream>>>(bih_w, bhh_w, bw, 4 * RNN);

  // ---- recurrence ----
  for (int t = 0; t < TSTEPS; ++t) {
    const float* wt   = emb + (size_t)t * BATCH * RNN;
    float* hr_in  = (t & 1) ? hrB : hrA;
    float* hr_out = (t & 1) ? hrA : hrB;
    float* hw_in  = (t & 1) ? hwB : hwA;
    float* hw_out = (t & 1) ? hwA : hwB;

    // read-LSTM: gates = [w_t | init_out] @ Wih_r^T + hr @ Whh_r^T + br
    build3_kernel<<<48, 256, 0, stream>>>(xr_h, wt, RNN, init_out, RNN, hr_in, RNN);
    gemm_kernel<<<128, 32, 0, stream>>>(xr_h, 1536,
                                        Wih_r_h, 1024, 1024,
                                        Whh_r_h, 512, 512,
                                        br, gates_ws, 4 * RNN);
    lstm_act_kernel<<<32, 256, 0, stream>>>(gates_ws, cr, cr, hr_out, nullptr);

    // memory attention: sim -> z -> m
    mem_read_kernel<<<KMEM * BATCH, 256, 0, stream>>>(Mq, hr_out, z_ws, m_ws);

    // comp = softmax([hr | m0 | m1] @ Wc^T + bc)
    build3_kernel<<<48, 256, 0, stream>>>(cat_h, hr_out, RNN,
                                          m_ws, RNN, m_ws + BATCH * RNN, RNN);
    gemm_kernel<<<32, 32, 0, stream>>>(cat_h, 1536,
                                       Wc_h, 1536, 1536,
                                       nullptr, 0, 0,
                                       bc, logit_ws, RNN);
    row_softmax_kernel<<<1, 512, 0, stream>>>(logit_ws, comp_ws);

    // write-LSTM: gates = comp @ Wih_w^T + hw @ Whh_w^T + bw
    build3_kernel<<<32, 256, 0, stream>>>(xw_h, comp_ws, RNN, hw_in, RNN,
                                          nullptr, 0);
    gemm_kernel<<<128, 32, 0, stream>>>(xw_h, 1024,
                                        Wih_w_h, 512, 512,
                                        Whh_w_h, 512, 512,
                                        bw, gates_ws, 4 * RNN);
    lstm_act_kernel<<<32, 256, 0, stream>>>(gates_ws, cw, cw, hw_out,
                                            out + (size_t)t * BATCH * RNN);

    // Mq = (1 - z) * (Mq + hw)
    mem_update_kernel<<<4096, 256, 0, stream>>>(Mq, z_ws, hw_out);
  }

  // ---- final states: outputs already written; t=63 -> h buffers end in A ----
  const size_t O_OUT = (size_t)TSTEPS * BATCH * RNN;   // 524288
  hipMemcpyAsync(out + O_OUT,                       hrA, BATCH * RNN * 4,
                 hipMemcpyDeviceToDevice, stream);
  hipMemcpyAsync(out + O_OUT + BATCH * RNN,         cr,  BATCH * RNN * 4,
                 hipMemcpyDeviceToDevice, stream);
  hipMemcpyAsync(out + O_OUT + 2 * BATCH * RNN,     hwA, BATCH * RNN * 4,
                 hipMemcpyDeviceToDevice, stream);
  hipMemcpyAsync(out + O_OUT + 3 * BATCH * RNN,     cw,  BATCH * RNN * 4,
                 hipMemcpyDeviceToDevice, stream);
  hipMemcpyAsync(out + O_OUT + 4 * BATCH * RNN,     Mq,  MQ_ELEMS * 4,
                 hipMemcpyDeviceToDevice, stream);
}
